// SinkhornTransformerSeq2SeqBlock_69020124446899
// MI455X (gfx1250) — compile-verified, matrix operand order checked by reading, not run on hardware
//
#include <hip/hip_runtime.h>
#include <hip/hip_bf16.h>

// ---------------- problem constants ----------------
#define Bsz 4
#define LTc 2048
#define LEc 1024
#define Dc  1024
#define Hc  16
#define HDc 64
#define Mff 4096
#define BSc 128
#define NBc 16
#define NEGV (-1e9f)
#define SCALEV 0.125f   // 1/sqrt(64)

typedef __bf16 v16bf  __attribute__((ext_vector_type(16)));
typedef __bf16 bf16x8 __attribute__((ext_vector_type(8)));
typedef float  v8f    __attribute__((ext_vector_type(8)));
typedef float  f4     __attribute__((ext_vector_type(4)));

// ---------------- WMMA helpers ----------------
static __device__ inline v8f wmma_bf16(v16bf a, v16bf b, v8f c) {
  return __builtin_amdgcn_wmma_f32_16x16x32_bf16(false, a, false, b, (short)0, c, false, false);
}

// Lane's 16 fragment elements = two contiguous 8-element (16B) runs:
//   k in [8*half, 8*half+8) and [16+8*half, 16+8*half+8)   (ISA 7.12.2)
static __device__ inline v16bf frag_row_bf(const __bf16* t, int row, int ld, int kb, int lane) {
  const int half = lane >> 4, r = lane & 15;
  const __bf16* base = t + (size_t)(row + r) * ld + kb + half * 8;
  bf16x8 lo = *(const bf16x8*)base;
  bf16x8 hi = *(const bf16x8*)(base + 16);
  return __builtin_shufflevector(lo, hi, 0, 1, 2, 3, 4, 5, 6, 7,
                                 8, 9, 10, 11, 12, 13, 14, 15);
}
static __device__ inline v16bf frag_row_f32(const float* t, int row, int ld, int kb, int lane) {
  const int half = lane >> 4, r = lane & 15;
  const float* base = t + (size_t)(row + r) * ld + kb + half * 8;
  f4 x0 = *(const f4*)base;
  f4 x1 = *(const f4*)(base + 4);
  f4 x2 = *(const f4*)(base + 16);
  f4 x3 = *(const f4*)(base + 20);
  v16bf f;
#pragma unroll
  for (int e = 0; e < 4; ++e) {
    f[e]      = (__bf16)x0[e];
    f[e + 4]  = (__bf16)x1[e];
    f[e + 8]  = (__bf16)x2[e];
    f[e + 12] = (__bf16)x3[e];
  }
  return f;
}

// ---------------- fp32 -> bf16 convert ----------------
__global__ __launch_bounds__(256) void k_cvt(const float* __restrict__ in,
                                             __bf16* __restrict__ out, long n) {
  long i = (long)blockIdx.x * blockDim.x + threadIdx.x;
  long stride = (long)gridDim.x * blockDim.x;
  for (; i < n; i += stride) out[i] = (__bf16)in[i];
}

// fp32 [K,N] -> bf16 [N,K] (tiled transpose)
__global__ __launch_bounds__(256) void k_cvt_t(const float* __restrict__ in,
                                               __bf16* __restrict__ out,
                                               int K, int N) {
  __shared__ float tile[32][33];
  const int k0 = blockIdx.y * 32, n0 = blockIdx.x * 32;
  const int tx = threadIdx.x & 31, ty = threadIdx.x >> 5;
  for (int i = ty; i < 32; i += 8)
    tile[i][tx] = in[(size_t)(k0 + i) * N + n0 + tx];
  __syncthreads();
  for (int i = ty; i < 32; i += 8)
    out[(size_t)(n0 + i) * K + k0 + tx] = (__bf16)tile[tx][i];
}

// ---------------- LayerNorm (D=1024) fp32 -> bf16 ----------------
__global__ __launch_bounds__(256) void k_ln(const float* __restrict__ x,
                                            const float* __restrict__ g,
                                            const float* __restrict__ b,
                                            __bf16* __restrict__ out) {
  __shared__ float red[256];
  const int tid = threadIdx.x;
  const float* xr = x + (size_t)blockIdx.x * Dc;
  float v[4], s = 0.f;
#pragma unroll
  for (int i = 0; i < 4; ++i) { v[i] = xr[tid + i * 256]; s += v[i]; }
  red[tid] = s; __syncthreads();
  for (int st = 128; st > 0; st >>= 1) { if (tid < st) red[tid] += red[tid + st]; __syncthreads(); }
  float mu = red[0] * (1.f / Dc);
  __syncthreads();
  s = 0.f;
#pragma unroll
  for (int i = 0; i < 4; ++i) { float d = v[i] - mu; s += d * d; }
  red[tid] = s; __syncthreads();
  for (int st = 128; st > 0; st >>= 1) { if (tid < st) red[tid] += red[tid + st]; __syncthreads(); }
  float rstd = rsqrtf(red[0] * (1.f / Dc) + 1e-6f);
  __bf16* o = out + (size_t)blockIdx.x * Dc;
#pragma unroll
  for (int i = 0; i < 4; ++i) {
    int c = tid + i * 256;
    o[c] = (__bf16)((v[i] - mu) * rstd * g[c] + b[c]);
  }
}

// ---------------- LDS-free WMMA GEMM ----------------
// C[M,N] = A[M,K](bf16,row) x Wt[N,K](bf16, pre-transposed)
// block = 128x128 (8 waves, each 32x64 = 2x4 16x16 tiles)
static __device__ inline float gelu_f(float x) {
  float x3 = x * x * x;
  return 0.5f * x * (1.f + tanhf(0.7978845608028654f * (x + 0.044715f * x3)));
}

__global__ __launch_bounds__(256)
void k_gemm(const __bf16* __restrict__ A, const __bf16* __restrict__ Wt,
            int M, int N, int K,
            const float* __restrict__ bias, const float* __restrict__ resid,
            int act_gelu, float* __restrict__ outF, __bf16* __restrict__ outB) {
  const int tid = threadIdx.x, lane = tid & 31, wid = tid >> 5;
  const int m0 = blockIdx.y * 128 + (wid >> 1) * 32;
  const int n0 = blockIdx.x * 128 + (wid & 1) * 64;
  const __bf16* Ab = A + (size_t)m0 * K;
  const __bf16* Bb = Wt + (size_t)n0 * K;
  v8f acc[2][4] = {};
  for (int kt = 0; kt < K; kt += 32) {
    v16bf a0 = frag_row_bf(Ab, 0, K, kt, lane);
    v16bf a1 = frag_row_bf(Ab, 16, K, kt, lane);
    v16bf b0 = frag_row_bf(Bb, 0, K, kt, lane);
    v16bf b1 = frag_row_bf(Bb, 16, K, kt, lane);
    v16bf b2 = frag_row_bf(Bb, 32, K, kt, lane);
    v16bf b3 = frag_row_bf(Bb, 48, K, kt, lane);
    acc[0][0] = wmma_bf16(a0, b0, acc[0][0]);
    acc[0][1] = wmma_bf16(a0, b1, acc[0][1]);
    acc[0][2] = wmma_bf16(a0, b2, acc[0][2]);
    acc[0][3] = wmma_bf16(a0, b3, acc[0][3]);
    acc[1][0] = wmma_bf16(a1, b0, acc[1][0]);
    acc[1][1] = wmma_bf16(a1, b1, acc[1][1]);
    acc[1][2] = wmma_bf16(a1, b2, acc[1][2]);
    acc[1][3] = wmma_bf16(a1, b3, acc[1][3]);
  }
  const int half = lane >> 4, nl = lane & 15;
#pragma unroll
  for (int i = 0; i < 2; ++i)
#pragma unroll
    for (int j = 0; j < 4; ++j)
#pragma unroll
      for (int r = 0; r < 8; ++r) {
        int m = m0 + i * 16 + r + half * 8;
        int n = n0 + j * 16 + nl;
        float v = acc[i][j][r];
        if (bias) v += bias[n];
        if (act_gelu) v = gelu_f(v);
        if (resid) v += resid[(size_t)m * N + n];
        size_t idx = (size_t)m * N + n;
        if (outF) outF[idx] = v;
        if (outB) outB[idx] = (__bf16)v;
      }
}

// ---------------- krep: sum over block rows ----------------
__global__ __launch_bounds__(64) void k_krep(const __bf16* __restrict__ k1,
                                             float* __restrict__ krep) {
  int d = threadIdx.x;
  int h = blockIdx.x % Hc, n = (blockIdx.x / Hc) % NBc, b = blockIdx.x / (Hc * NBc);
  const __bf16* base = k1 + ((size_t)(b * LTc + n * BSc) * Hc + h) * HDc + d;
  float s = 0.f;
  for (int ss = 0; ss < BSc; ++ss) s += (float)base[(size_t)ss * Dc];
  krep[((size_t)((b * NBc + n) * Hc + h)) * HDc + d] = s;
}

// ---------------- sort logits: [16x64] x [64x16] per (b,h) ----------------
__global__ __launch_bounds__(256) void k_logits(const float* __restrict__ krep,
                                                const float* __restrict__ wsort,
                                                float* __restrict__ logits) {
  int b = blockIdx.x / Hc, h = blockIdx.x % Hc;
  int n = threadIdx.x >> 4, mcol = threadIdx.x & 15;
  const float* kr = krep + ((size_t)(b * NBc + n) * Hc + h) * HDc;
  const float* ws = wsort + (size_t)h * HDc * NBc + mcol;
  float s = 0.f;
  for (int d = 0; d < HDc; ++d) s += kr[d] * ws[(size_t)d * NBc];
  logits[((size_t)(b * Hc + h) * NBc + n) * NBc + mcol] = s;
}

// ---------------- causal sinkhorn, 5 iters, 16x16 per (b,h) ----------------
__global__ __launch_bounds__(256) void k_sinkhorn(const float* __restrict__ logits,
                                                  float* __restrict__ P) {
  __shared__ float la[256];
  __shared__ float lse[16];
  const int tid = threadIdx.x;
  const int n = tid >> 4, mc = tid & 15;
  const bool allow = (mc <= n);
  la[tid] = allow ? logits[(size_t)blockIdx.x * 256 + tid] : NEGV;
  __syncthreads();
  for (int it = 0; it < 5; ++it) {
    if (mc == 0) {
      float mx = -1e30f;
      for (int j = 0; j < 16; ++j) mx = fmaxf(mx, la[n * 16 + j]);
      float s = 0.f;
      for (int j = 0; j < 16; ++j) s += __expf(la[n * 16 + j] - mx);
      lse[n] = mx + __logf(s);
    }
    __syncthreads();
    la[tid] = allow ? (la[tid] - lse[n]) : NEGV;
    __syncthreads();
    if (n == 0) {
      float mx = -1e30f;
      for (int j = 0; j < 16; ++j) mx = fmaxf(mx, la[j * 16 + mc]);
      float s = 0.f;
      for (int j = 0; j < 16; ++j) s += __expf(la[j * 16 + mc] - mx);
      lse[mc] = mx + __logf(s);
    }
    __syncthreads();
    la[tid] = allow ? (la[tid] - lse[mc]) : NEGV;
    __syncthreads();
  }
  P[(size_t)blockIdx.x * 256 + tid] = __expf(la[tid]);
}

// ---------------- ks/vs = P-weighted mix of key/value blocks ----------------
__global__ __launch_bounds__(256) void k_mix(const float* __restrict__ P,
                                             const __bf16* __restrict__ k1,
                                             const __bf16* __restrict__ v1,
                                             __bf16* __restrict__ kso,
                                             __bf16* __restrict__ vso) {
  int s = blockIdx.x % BSc, i = (blockIdx.x / BSc) % NBc, b = blockIdx.x / (BSc * NBc);
  int hd0 = threadIdx.x * 4, h = hd0 >> 6;
  const float* Pr = P + ((size_t)(b * Hc + h) * NBc + i) * NBc;
  float ak[4] = {}, av[4] = {};
  for (int j = 0; j < NBc; ++j) {
    float p = Pr[j];
    const __bf16* kb = k1 + (size_t)(b * LTc + j * BSc + s) * Dc + hd0;
    const __bf16* vb = v1 + (size_t)(b * LTc + j * BSc + s) * Dc + hd0;
#pragma unroll
    for (int t = 0; t < 4; ++t) { ak[t] += p * (float)kb[t]; av[t] += p * (float)vb[t]; }
  }
  __bf16* ko = kso + (size_t)(b * LTc + i * BSc + s) * Dc + hd0;
  __bf16* vo = vso + (size_t)(b * LTc + i * BSc + s) * Dc + hd0;
#pragma unroll
  for (int t = 0; t < 4; ++t) { ko[t] = (__bf16)ak[t]; vo[t] = (__bf16)av[t]; }
}

// ---------------- streaming WMMA attention ----------------
// mode 0: self (qb = n*2+qt; kv = 2 causal local chunks + 2 sorted chunks)
// mode 1: cross (kv = LE/64 chunks, no mask)
__global__ __launch_bounds__(256)
void k_attn(const __bf16* __restrict__ Q, const __bf16* __restrict__ Kl,
            const __bf16* __restrict__ Vl, const __bf16* __restrict__ Ksrt,
            const __bf16* __restrict__ Vsrt, __bf16* __restrict__ Out, int mode) {
  __shared__ __bf16 Qs[64 * 72], Ks[64 * 72], VsT[64 * 72];  // ld 72 (144B rows)
  __shared__ float Ss[64 * 68], Os[64 * 68];                 // ld 68
  __shared__ float mrow[64], lrow[64], arow[64];
  const int tid = threadIdx.x, lane = tid & 31, wid = tid >> 5;
  const int qb = blockIdx.x, h = blockIdx.y, b = blockIdx.z;
  const int halfl = lane >> 4, nl = lane & 15;
  const size_t qrow0 = (size_t)b * LTc + qb * 64;

  { // load Q tile 64x64
    int row = tid >> 2, cg = (tid & 3) * 16;
    const bf16x8* src = (const bf16x8*)(Q + (qrow0 + row) * Dc + h * HDc + cg);
    bf16x8* dst = (bf16x8*)(Qs + row * 72 + cg);
    dst[0] = src[0]; dst[1] = src[1];
  }
  if (tid < 64) { mrow[tid] = -1e30f; lrow[tid] = 0.f; }
  { int row = tid >> 2, cg = (tid & 3) * 16;
    for (int j = 0; j < 16; ++j) Os[row * 68 + cg + j] = 0.f; }
  __syncthreads();

  const int nchunks = mode ? (LEc / 64) : 4;
  for (int c = 0; c < nchunks; ++c) {
    const __bf16 *kc, *vc;
    int causal = 0, kbase = 0, qbase = 0;
    if (mode == 0) {
      int nblk = qb >> 1, qt = qb & 1;
      size_t krow0 = (size_t)b * LTc + nblk * BSc + (c & 1) * 64;
      if (c < 2) {
        kc = Kl + krow0 * Dc + h * HDc; vc = Vl + krow0 * Dc + h * HDc;
        causal = 1; kbase = c * 64; qbase = qt * 64;
      } else {
        kc = Ksrt + krow0 * Dc + h * HDc; vc = Vsrt + krow0 * Dc + h * HDc;
      }
    } else {
      size_t krow0 = (size_t)b * LEc + c * 64;
      kc = Kl + krow0 * Dc + h * HDc; vc = Vl + krow0 * Dc + h * HDc;
    }
    { // load K chunk (row major) and V chunk (transposed to [d][s])
      int row = tid >> 2, cg = (tid & 3) * 16;
      const bf16x8* sk = (const bf16x8*)(kc + (size_t)row * Dc + cg);
      bf16x8* dk = (bf16x8*)(Ks + row * 72 + cg);
      dk[0] = sk[0]; dk[1] = sk[1];
      bf16x8 sv0 = *(const bf16x8*)(vc + (size_t)row * Dc + cg);
      bf16x8 sv1 = *(const bf16x8*)(vc + (size_t)row * Dc + cg + 8);
#pragma unroll
      for (int j = 0; j < 8; ++j) {
        VsT[(cg + j) * 72 + row] = sv0[j];
        VsT[(cg + 8 + j) * 72 + row] = sv1[j];
      }
    }
    __syncthreads();
    // S = Q K^T (64x64), 16 tiles, 2 per wave
#pragma unroll
    for (int tt = 0; tt < 2; ++tt) {
      int t = wid * 2 + tt, ti = t >> 2, tj = t & 3;
      v8f s8 = {};
#pragma unroll
      for (int kb2 = 0; kb2 < 64; kb2 += 32) {
        v16bf af = frag_row_bf(Qs, ti * 16, 72, kb2, lane);
        v16bf bfv = frag_row_bf(Ks, tj * 16, 72, kb2, lane);
        s8 = wmma_bf16(af, bfv, s8);
      }
#pragma unroll
      for (int r = 0; r < 8; ++r) {
        int mq = ti * 16 + r + halfl * 8;
        int nk = tj * 16 + nl;
        float v = s8[r] * SCALEV;
        if (causal && (kbase + nk > qbase + mq)) v = NEGV;
        Ss[mq * 68 + nk] = v;
      }
    }
    __syncthreads();
    // online softmax per row
    if (tid < 64) {
      float mold = mrow[tid], mx = mold;
      for (int j = 0; j < 64; ++j) mx = fmaxf(mx, Ss[tid * 68 + j]);
      float al = __expf(mold - mx);
      float s = 0.f;
      for (int j = 0; j < 64; ++j) {
        float pv = __expf(Ss[tid * 68 + j] - mx);
        Ss[tid * 68 + j] = pv; s += pv;
      }
      lrow[tid] = lrow[tid] * al + s; mrow[tid] = mx; arow[tid] = al;
    }
    __syncthreads();
    { // rescale O accumulator
      int row = tid >> 2, cg = (tid & 3) * 16;
      float al = arow[row];
      for (int j = 0; j < 16; ++j) Os[row * 68 + cg + j] *= al;
    }
    __syncthreads();
    // O += P x V (64x64): A = P rows (fp32->bf16), B = VsT rows ([d][s])
#pragma unroll
    for (int tt = 0; tt < 2; ++tt) {
      int t = wid * 2 + tt, ti = t >> 2, tj = t & 3;
      v8f c8;
#pragma unroll
      for (int r = 0; r < 8; ++r)
        c8[r] = Os[(ti * 16 + r + halfl * 8) * 68 + tj * 16 + nl];
#pragma unroll
      for (int kb2 = 0; kb2 < 64; kb2 += 32) {
        v16bf af = frag_row_f32(Ss, ti * 16, 68, kb2, lane);
        v16bf bfv = frag_row_bf(VsT, tj * 16, 72, kb2, lane);
        c8 = wmma_bf16(af, bfv, c8);
      }
#pragma unroll
      for (int r = 0; r < 8; ++r)
        Os[(ti * 16 + r + halfl * 8) * 68 + tj * 16 + nl] = c8[r];
    }
    __syncthreads();
  }
  { // normalize + write bf16
    int row = tid >> 2, cg = (tid & 3) * 16;
    float inv = 1.f / lrow[row];
    __bf16* dst = Out + (qrow0 + row) * Dc + h * HDc + cg;
    for (int j = 0; j < 16; ++j) dst[j] = (__bf16)(Os[row * 68 + cg + j] * inv);
  }
}

// ---------------- host-side orchestration ----------------
extern "C" void kernel_launch(void* const* d_in, const int* in_sizes, int n_in,
                              void* d_out, int out_size, void* d_ws, size_t ws_size,
                              hipStream_t stream) {
  const float* targets = (const float*)d_in[0];
  const float* encoded = (const float*)d_in[1];
  const float* ln1_s = (const float*)d_in[2]; const float* ln1_b = (const float*)d_in[3];
  const float* ln2_s = (const float*)d_in[4]; const float* ln2_b = (const float*)d_in[5];
  const float* ln3_s = (const float*)d_in[6]; const float* ln3_b = (const float*)d_in[7];
  const float* wq1 = (const float*)d_in[8];  const float* wk1 = (const float*)d_in[9];
  const float* wv1 = (const float*)d_in[10]; const float* wo1 = (const float*)d_in[11];
  const float* wsort = (const float*)d_in[12];
  const float* wq2 = (const float*)d_in[13]; const float* wk2 = (const float*)d_in[14];
  const float* wv2 = (const float*)d_in[15]; const float* wo2 = (const float*)d_in[16];
  const float* w1 = (const float*)d_in[17];  const float* b1 = (const float*)d_in[18];
  const float* w2 = (const float*)d_in[19];  const float* b2 = (const float*)d_in[20];

  char* p = (char*)d_ws;
  auto alloc = [&](size_t bytes) { char* r = p; p += (bytes + 255) & ~(size_t)255; return r; };
  const size_t DD = (size_t)Dc * Dc;          // 1M
  const size_t ROWS = (size_t)Bsz * LTc;      // 8192
  const size_t EROWS = (size_t)Bsz * LEc;     // 4096

  // weights stored TRANSPOSED as [N,K] bf16
  __bf16* wq1t = (__bf16*)alloc(DD * 2);
  __bf16* wk1t = (__bf16*)alloc(DD * 2);
  __bf16* wv1t = (__bf16*)alloc(DD * 2);
  __bf16* wo1t = (__bf16*)alloc(DD * 2);
  __bf16* wq2t = (__bf16*)alloc(DD * 2);
  __bf16* wk2t = (__bf16*)alloc(DD * 2);
  __bf16* wv2t = (__bf16*)alloc(DD * 2);
  __bf16* wo2t = (__bf16*)alloc(DD * 2);
  __bf16* w1t  = (__bf16*)alloc((size_t)Dc * Mff * 2);
  __bf16* w2t  = (__bf16*)alloc((size_t)Mff * Dc * 2);
  __bf16* encb = (__bf16*)alloc(EROWS * Dc * 2);
  __bf16* xln1 = (__bf16*)alloc(ROWS * Dc * 2);
  __bf16* q1   = (__bf16*)alloc(ROWS * Dc * 2);
  __bf16* k1   = (__bf16*)alloc(ROWS * Dc * 2);
  __bf16* v1   = (__bf16*)alloc(ROWS * Dc * 2);
  float*  krep = (float*)alloc((size_t)Bsz * NBc * Hc * HDc * 4);
  float*  lgts = (float*)alloc((size_t)Bsz * Hc * NBc * NBc * 4);
  float*  Pm   = (float*)alloc((size_t)Bsz * Hc * NBc * NBc * 4);
  __bf16* kso  = (__bf16*)alloc(ROWS * Dc * 2);
  __bf16* vso  = (__bf16*)alloc(ROWS * Dc * 2);
  __bf16* att1 = (__bf16*)alloc(ROWS * Dc * 2);
  float*  x    = (float*)alloc(ROWS * Dc * 4);
  __bf16* yln  = (__bf16*)alloc(ROWS * Dc * 2);
  __bf16* q2   = (__bf16*)alloc(ROWS * Dc * 2);
  __bf16* k2   = (__bf16*)alloc(EROWS * Dc * 2);
  __bf16* v2   = (__bf16*)alloc(EROWS * Dc * 2);
  __bf16* att2 = (__bf16*)alloc(ROWS * Dc * 2);
  float*  y    = (float*)alloc(ROWS * Dc * 4);
  __bf16* zln  = (__bf16*)alloc(ROWS * Dc * 2);
  __bf16* ffh  = (__bf16*)alloc(ROWS * Mff * 2);
  float*  outF = (float*)d_out;

  // --- weight transpose-conversions to bf16 [N,K] ---
  dim3 gT(Dc / 32, Dc / 32);
  k_cvt_t<<<gT, 256, 0, stream>>>(wq1, wq1t, Dc, Dc);
  k_cvt_t<<<gT, 256, 0, stream>>>(wk1, wk1t, Dc, Dc);
  k_cvt_t<<<gT, 256, 0, stream>>>(wv1, wv1t, Dc, Dc);
  k_cvt_t<<<gT, 256, 0, stream>>>(wo1, wo1t, Dc, Dc);
  k_cvt_t<<<gT, 256, 0, stream>>>(wq2, wq2t, Dc, Dc);
  k_cvt_t<<<gT, 256, 0, stream>>>(wk2, wk2t, Dc, Dc);
  k_cvt_t<<<gT, 256, 0, stream>>>(wv2, wv2t, Dc, Dc);
  k_cvt_t<<<gT, 256, 0, stream>>>(wo2, wo2t, Dc, Dc);
  dim3 gT1(Mff / 32, Dc / 32);   // w1: [Dc,Mff] -> [Mff,Dc]
  k_cvt_t<<<gT1, 256, 0, stream>>>(w1, w1t, Dc, Mff);
  dim3 gT2(Dc / 32, Mff / 32);   // w2: [Mff,Dc] -> [Dc,Mff]
  k_cvt_t<<<gT2, 256, 0, stream>>>(w2, w2t, Mff, Dc);
  k_cvt<<<2048, 256, 0, stream>>>(encoded, encb, (long)EROWS * Dc);

  // --- LN1 + QKV projections ---
  k_ln<<<ROWS, 256, 0, stream>>>(targets, ln1_s, ln1_b, xln1);
  dim3 gP(Dc / 128, ROWS / 128);
  k_gemm<<<gP, 256, 0, stream>>>(xln1, wq1t, (int)ROWS, Dc, Dc, nullptr, nullptr, 0, nullptr, q1);
  k_gemm<<<gP, 256, 0, stream>>>(xln1, wk1t, (int)ROWS, Dc, Dc, nullptr, nullptr, 0, nullptr, k1);
  k_gemm<<<gP, 256, 0, stream>>>(xln1, wv1t, (int)ROWS, Dc, Dc, nullptr, nullptr, 0, nullptr, v1);

  // --- SortNet + sinkhorn + block mixing ---
  k_krep<<<Bsz * NBc * Hc, 64, 0, stream>>>(k1, krep);
  k_logits<<<Bsz * Hc, 256, 0, stream>>>(krep, wsort, lgts);
  k_sinkhorn<<<Bsz * Hc, 256, 0, stream>>>(lgts, Pm);
  k_mix<<<Bsz * NBc * BSc, 256, 0, stream>>>(Pm, k1, v1, kso, vso);

  // --- self attention + WO1 (+ residual targets) ---
  dim3 gA(LTc / 64, Hc, Bsz);
  k_attn<<<gA, 256, 0, stream>>>(q1, k1, v1, kso, vso, att1, 0);
  k_gemm<<<gP, 256, 0, stream>>>(att1, wo1t, (int)ROWS, Dc, Dc, nullptr, targets, 0, x, nullptr);

  // --- LN2 + cross attention + WO2 (+ residual x) ---
  k_ln<<<ROWS, 256, 0, stream>>>(x, ln2_s, ln2_b, yln);
  k_gemm<<<gP, 256, 0, stream>>>(yln, wq2t, (int)ROWS, Dc, Dc, nullptr, nullptr, 0, nullptr, q2);
  dim3 gE(Dc / 128, EROWS / 128);
  k_gemm<<<gE, 256, 0, stream>>>(encb, wk2t, (int)EROWS, Dc, Dc, nullptr, nullptr, 0, nullptr, k2);
  k_gemm<<<gE, 256, 0, stream>>>(encb, wv2t, (int)EROWS, Dc, Dc, nullptr, nullptr, 0, nullptr, v2);
  k_attn<<<gA, 256, 0, stream>>>(q2, k2, v2, nullptr, nullptr, att2, 1);
  k_gemm<<<gP, 256, 0, stream>>>(att2, wo2t, (int)ROWS, Dc, Dc, nullptr, x, 0, y, nullptr);

  // --- LN3 + FFN (+ residual y) -> d_out ---
  k_ln<<<ROWS, 256, 0, stream>>>(y, ln3_s, ln3_b, zln);
  dim3 gF1(Mff / 128, ROWS / 128);
  k_gemm<<<gF1, 256, 0, stream>>>(zln, w1t, (int)ROWS, Mff, Dc, b1, nullptr, 1, nullptr, ffh);
  dim3 gF2(Dc / 128, ROWS / 128);
  k_gemm<<<gF2, 256, 0, stream>>>(ffh, w2t, (int)ROWS, Dc, Mff, b2, y, 0, outF, nullptr);
}
